// CasualSelfAttention_alternative_53317724012591
// MI455X (gfx1250) — matrix operand order repaired and checked by reading, not compiled, not run on hardware
//
#include <hip/hip_runtime.h>
#include <hip/hip_bf16.h>
#include <math.h>

typedef unsigned short u16;
typedef __attribute__((ext_vector_type(16))) __bf16 v16bf;
typedef __attribute__((ext_vector_type(8)))  float  v8f;

#define B_  4
#define S_  2048
#define E_  1024
#define H_  16
#define HD_ 64
#define M_  (B_ * S_)   // 8192 rows

// ---------- helpers ----------

__device__ __forceinline__ u16 f2bf(float f) {
  unsigned int u = __float_as_uint(f);
  u += 0x7FFFu + ((u >> 16) & 1u);      // round-to-nearest-even
  return (u16)(u >> 16);
}

union FragU {
  v16bf v;
  uint4 u[2];
};

// A-operand layout (16-bit A 16x32, wave32):
// lane half hi: VGPR0..3 = K (k0+hi*8)..+7 ; VGPR4..7 = K (k0+16+hi*8)..+7
__device__ __forceinline__ v16bf load_frag_a(const u16* base, int ld, int row, int k0, int hi) {
  FragU f;
  const u16* p = base + (size_t)row * ld + k0 + hi * 8;
  f.u[0] = *(const uint4*)(p);
  f.u[1] = *(const uint4*)(p + 16);
  return f.v;
}

// B-operand layout (16-bit B 32x16, wave32):
// lane half hi: VGPR0..3 = K (k0+hi*8)..+7 ; VGPR4..7 = K (k0+16+hi*8)..+7
__device__ __forceinline__ v16bf load_frag_b(const u16* base, int ld, int row, int k0, int hi) {
  FragU f;
  const u16* p = base + (size_t)row * ld + k0 + hi * 8;
  f.u[0] = *(const uint4*)(p);
  f.u[1] = *(const uint4*)(p + 16);
  return f.v;
}

// ---------- elementwise / transpose kernels ----------

__global__ void cvt_f32_bf16(const float* __restrict__ in, u16* __restrict__ out, int n) {
  int i = blockIdx.x * blockDim.x + threadIdx.x;
  if (i < n) out[i] = f2bf(in[i]);
}

// W[k][n] f32 -> Wt[n][k] bf16 (so B-operand rows are contiguous over k)
__global__ void transpose_w(const float* __restrict__ W, u16* __restrict__ Wt) {
  int i = blockIdx.x * blockDim.x + threadIdx.x;  // E*E
  int k = i & (E_ - 1);
  int n = i >> 10;
  Wt[i] = f2bf(W[(size_t)k * E_ + n]);
}

// V bf16 [B,S,E] -> Vt bf16 [B,H,HD,S]  (PV needs V^T per head)
__global__ void transpose_v(const u16* __restrict__ V, u16* __restrict__ Vt) {
  int i = blockIdx.x * blockDim.x + threadIdx.x;  // B*H*HD*S
  int s  = i & (S_ - 1);
  int r  = i >> 11;               // (b*H + h)*HD + d
  int d  = r & (HD_ - 1);
  int bh = r >> 6;
  int h  = bh & (H_ - 1);
  int b  = bh >> 4;
  Vt[i] = V[((size_t)(b * S_ + s)) * E_ + h * HD_ + d];
}

// ---------- bf16 WMMA GEMM:  C[M,N] = A[M,K] @ Bt[N,K]^T + bias ----------
// block = 256 threads = 8 waves arranged 4(M) x 2(N); wave tile = 32x64; block tile = 128x128

template <bool OUT_BF16>
__global__ __launch_bounds__(256) void gemm_bf16(const u16* __restrict__ A,
                                                 const u16* __restrict__ Bt,
                                                 const float* __restrict__ bias,
                                                 float* __restrict__ Cf,
                                                 u16* __restrict__ Cb,
                                                 int M, int N, int K) {
  const int lane = threadIdx.x & 31;
  const int wave = threadIdx.x >> 5;
  const int l = lane & 15, hi = lane >> 4;
  const int m0 = blockIdx.x * 128 + (wave & 3) * 32;
  const int n0 = blockIdx.y * 128 + (wave >> 2) * 64;

  const v8f vzero = {0.f, 0.f, 0.f, 0.f, 0.f, 0.f, 0.f, 0.f};
  v8f acc[2][4];
#pragma unroll
  for (int mt = 0; mt < 2; ++mt)
#pragma unroll
    for (int nt = 0; nt < 4; ++nt) acc[mt][nt] = vzero;

  for (int k0 = 0; k0 < K; k0 += 32) {
    v16bf a[2], b[4];
#pragma unroll
    for (int mt = 0; mt < 2; ++mt) a[mt] = load_frag_a(A, K, m0 + mt * 16 + l, k0, hi);
#pragma unroll
    for (int nt = 0; nt < 4; ++nt) b[nt] = load_frag_b(Bt, K, n0 + nt * 16 + l, k0, hi);
#pragma unroll
    for (int mt = 0; mt < 2; ++mt)
#pragma unroll
      for (int nt = 0; nt < 4; ++nt)
        acc[mt][nt] = __builtin_amdgcn_wmma_f32_16x16x32_bf16(
            false, a[mt], false, b[nt], (short)0, acc[mt][nt], false, false);
  }

  float bv[4];
#pragma unroll
  for (int nt = 0; nt < 4; ++nt) bv[nt] = bias[n0 + nt * 16 + l];

#pragma unroll
  for (int mt = 0; mt < 2; ++mt)
#pragma unroll
    for (int nt = 0; nt < 4; ++nt)
#pragma unroll
      for (int v = 0; v < 8; ++v) {
        const int row = m0 + mt * 16 + v + 8 * hi;   // C layout: VGPR v -> M = v + 8*hi
        const int col = n0 + nt * 16 + l;            // N = lane%16
        const float val = acc[mt][nt][v] + bv[nt];
        if (OUT_BF16) Cb[(size_t)row * N + col] = f2bf(val);
        else          Cf[(size_t)row * N + col] = val;
      }
}

// ---------- flash attention (causal, online softmax), wave = 32 query rows ----------

__global__ __launch_bounds__(256) void flash_attn(const u16* __restrict__ Q,
                                                  const u16* __restrict__ Kv,
                                                  const u16* __restrict__ Vt,
                                                  u16* __restrict__ Y) {
  const int lane = threadIdx.x & 31;
  const int wave = threadIdx.x >> 5;
  const int l = lane & 15, hi = lane >> 4;

  const int nqc = S_ / 256;                 // q-chunks per (b,h)
  const int qc = blockIdx.x % nqc;
  const int h  = (blockIdx.x / nqc) % H_;
  const int b  = blockIdx.x / (nqc * H_);
  const int q0 = qc * 256 + wave * 32;

  const u16* Qb = Q  + (size_t)b * S_ * E_ + h * HD_;          // rows = q pos, ld = E
  const u16* Kb = Kv + (size_t)b * S_ * E_ + h * HD_;          // rows = k pos, ld = E
  const u16* Vb = Vt + ((size_t)(b * H_ + h)) * HD_ * S_;      // rows = d,     ld = S

  __shared__ u16 pshm[8][32 * 64];          // per-wave 32x64 P staging (C->A relayout)
  u16* P = pshm[wave];

  const v8f vzero = {0.f, 0.f, 0.f, 0.f, 0.f, 0.f, 0.f, 0.f};
  v8f o[2][4];
#pragma unroll
  for (int mt = 0; mt < 2; ++mt)
#pragma unroll
    for (int dt = 0; dt < 4; ++dt) o[mt][dt] = vzero;

  float rmax[2][8], rsum[2][8];
#pragma unroll
  for (int mt = 0; mt < 2; ++mt)
#pragma unroll
    for (int v = 0; v < 8; ++v) { rmax[mt][v] = -1.0e30f; rsum[mt][v] = 0.f; }

  // Q fragments are loop-invariant: hoist (2 m-tiles x 2 k-steps over HD=64)
  v16bf qa[2][2];
#pragma unroll
  for (int mt = 0; mt < 2; ++mt)
#pragma unroll
    for (int kk = 0; kk < 2; ++kk)
      qa[mt][kk] = load_frag_a(Qb, E_, q0 + mt * 16 + l, kk * 32, hi);

  const float scale = 0.125f;               // 1/sqrt(64)

  for (int j = 0; j < q0 + 32; j += 64) {   // causal: only key tiles <= query tile
    // ---- S = Q K^T (32x64 tile) ----
    v8f sc[2][4];
#pragma unroll
    for (int mt = 0; mt < 2; ++mt)
#pragma unroll
      for (int nt = 0; nt < 4; ++nt) sc[mt][nt] = vzero;

#pragma unroll
    for (int kk = 0; kk < 2; ++kk) {
      v16bf kb[4];
#pragma unroll
      for (int nt = 0; nt < 4; ++nt)
        kb[nt] = load_frag_b(Kb, E_, j + nt * 16 + l, kk * 32, hi);
#pragma unroll
      for (int mt = 0; mt < 2; ++mt)
#pragma unroll
        for (int nt = 0; nt < 4; ++nt)
          sc[mt][nt] = __builtin_amdgcn_wmma_f32_16x16x32_bf16(
              false, qa[mt][kk], false, kb[nt], (short)0, sc[mt][nt], false, false);
    }

    // ---- causal mask + online softmax (per row = (mt, v, hi)) ----
#pragma unroll
    for (int mt = 0; mt < 2; ++mt)
#pragma unroll
      for (int v = 0; v < 8; ++v) {
        const int qrow = q0 + mt * 16 + v + 8 * hi;
        float mx = -1.0e30f;
#pragma unroll
        for (int nt = 0; nt < 4; ++nt) {
          const int kcol = j + nt * 16 + l;
          float x = sc[mt][nt][v] * scale;
          x = (kcol <= qrow) ? x : -1.0e30f;
          sc[mt][nt][v] = x;
          mx = fmaxf(mx, x);
        }
        // reduce across the 16 lanes of this half (xor masks 1..8 keep halves separate)
#pragma unroll
        for (int off = 1; off < 16; off <<= 1) mx = fmaxf(mx, __shfl_xor(mx, off, 32));
        const float mold = rmax[mt][v];
        const float mnew = fmaxf(mold, mx);
        const float alpha = __expf(mold - mnew);
        rmax[mt][v] = mnew;
        float ps = 0.f;
#pragma unroll
        for (int nt = 0; nt < 4; ++nt) {
          const float p = __expf(sc[mt][nt][v] - mnew);
          sc[mt][nt][v] = p;
          ps += p;
        }
#pragma unroll
        for (int off = 1; off < 16; off <<= 1) ps += __shfl_xor(ps, off, 32);
        rsum[mt][v] = rsum[mt][v] * alpha + ps;
#pragma unroll
        for (int dt = 0; dt < 4; ++dt) o[mt][dt][v] *= alpha;   // rescale running O rows
      }

    // ---- C-layout -> A-layout via per-wave LDS tile (bf16) ----
#pragma unroll
    for (int mt = 0; mt < 2; ++mt)
#pragma unroll
      for (int nt = 0; nt < 4; ++nt)
#pragma unroll
        for (int v = 0; v < 8; ++v)
          P[(mt * 16 + v + 8 * hi) * 64 + nt * 16 + l] = f2bf(sc[mt][nt][v]);

    // ---- O += P @ V (contraction over 64 key positions of this tile) ----
#pragma unroll
    for (int kk = 0; kk < 2; ++kk) {
      v16bf pa[2], vb[4];
#pragma unroll
      for (int mt = 0; mt < 2; ++mt)
        pa[mt] = load_frag_a(P, 64, mt * 16 + l, kk * 32, hi);     // ds_load_b128
#pragma unroll
      for (int dt = 0; dt < 4; ++dt)
        vb[dt] = load_frag_b(Vb, S_, dt * 16 + l, j + kk * 32, hi);
#pragma unroll
      for (int mt = 0; mt < 2; ++mt)
#pragma unroll
        for (int dt = 0; dt < 4; ++dt)
          o[mt][dt] = __builtin_amdgcn_wmma_f32_16x16x32_bf16(
              false, pa[mt], false, vb[dt], (short)0, o[mt][dt], false, false);
    }
  }

  // ---- epilogue: O / rowsum -> Y bf16 [B,S,E] (heads merge contiguously) ----
#pragma unroll
  for (int mt = 0; mt < 2; ++mt)
#pragma unroll
    for (int dt = 0; dt < 4; ++dt)
#pragma unroll
      for (int v = 0; v < 8; ++v) {
        const int row = q0 + mt * 16 + v + 8 * hi;
        const int col = h * HD_ + dt * 16 + l;
        Y[((size_t)(b * S_ + row)) * E_ + col] = f2bf(o[mt][dt][v] / rsum[mt][v]);
      }
}

// ---------- launcher ----------

extern "C" void kernel_launch(void* const* d_in, const int* in_sizes, int n_in,
                              void* d_out, int out_size, void* d_ws, size_t ws_size,
                              hipStream_t stream) {
  const float* x  = (const float*)d_in[0];
  const float* Wq = (const float*)d_in[1];
  const float* bq = (const float*)d_in[2];
  const float* Wk = (const float*)d_in[3];
  const float* bk = (const float*)d_in[4];
  const float* Wv = (const float*)d_in[5];
  const float* bv = (const float*)d_in[6];
  const float* Wo = (const float*)d_in[7];
  const float* bo = (const float*)d_in[8];
  float* out = (float*)d_out;

  char* ws = (char*)d_ws;
  u16* xb  = (u16*)(ws);                        // 16 MB  x in bf16
  u16* WqT = (u16*)(ws + (16u << 20));          //  2 MB each, transposed bf16
  u16* WkT = (u16*)(ws + (18u << 20));
  u16* WvT = (u16*)(ws + (20u << 20));
  u16* WoT = (u16*)(ws + (22u << 20));
  u16* Qb  = (u16*)(ws + (24u << 20));          // 16 MB
  u16* Kb  = (u16*)(ws + (40u << 20));          // 16 MB
  u16* Vb  = (u16*)(ws + (56u << 20));          // 16 MB
  u16* Vtb = (u16*)(ws + (72u << 20));          // 16 MB
  u16* Yb  = (u16*)(ws + (88u << 20));          // 16 MB  (total 104 MB)

  {
    int n = M_ * E_;
    cvt_f32_bf16<<<(n + 255) / 256, 256, 0, stream>>>(x, xb, n);
  }
  {
    int n = E_ * E_;
    int g = (n + 255) / 256;
    transpose_w<<<g, 256, 0, stream>>>(Wq, WqT);
    transpose_w<<<g, 256, 0, stream>>>(Wk, WkT);
    transpose_w<<<g, 256, 0, stream>>>(Wv, WvT);
    transpose_w<<<g, 256, 0, stream>>>(Wo, WoT);
  }

  dim3 gg(M_ / 128, E_ / 128);
  gemm_bf16<true><<<gg, 256, 0, stream>>>(xb, WqT, bq, nullptr, Qb, M_, E_, E_);
  gemm_bf16<true><<<gg, 256, 0, stream>>>(xb, WkT, bk, nullptr, Kb, M_, E_, E_);
  gemm_bf16<true><<<gg, 256, 0, stream>>>(xb, WvT, bv, nullptr, Vb, M_, E_, E_);

  {
    int n = B_ * H_ * HD_ * S_;
    transpose_v<<<(n + 255) / 256, 256, 0, stream>>>(Vb, Vtb);
  }

  flash_attn<<<B_ * H_ * (S_ / 256), 256, 0, stream>>>(Qb, Kb, Vtb, Yb);

  gemm_bf16<false><<<gg, 256, 0, stream>>>(Yb, WoT, bo, out, nullptr, M_, E_, E_);
}
